// mHCBlock_28527172780141
// MI455X (gfx1250) — matrix operand order repaired and checked by reading, not compiled, not run on hardware
//
#include <hip/hip_runtime.h>
#include <hip/hip_bf16.h>

typedef __bf16 bf16;
typedef __attribute__((ext_vector_type(16))) __bf16 bf16x16;
typedef __attribute__((ext_vector_type(8)))  float  f32x8;
typedef unsigned int u32x4 __attribute__((ext_vector_type(4)));
typedef int          i32x4 __attribute__((ext_vector_type(4)));
typedef int          i32x8 __attribute__((ext_vector_type(8)));

#define PX_TOTAL 262144      // B*H*W
#define SC 256
#define SB 128               // pixels per block (kernel A)
#define XS 258               // padded x row stride (floats) in LDS (TDM pad: 256+2)
#define HS 34                // padded h staging stride (floats)
#define WS_CONV 584          // padded transposed weight stride (bf16)

#if defined(__has_builtin)
#if __has_builtin(__builtin_amdgcn_tensor_load_to_lds) && __has_builtin(__builtin_amdgcn_s_wait_tensorcnt)
#define HAVE_TDM 1
#endif
#endif

// ---------------------------------------------------------------------------
// Kernel A: RMSNorm + 24-way projection GEMM (bf16 WMMA) + sigmoid/Sinkhorn
//           + residual stream mix (writes out), x_pre (bf16) and h_post to ws.
//           x tile staged to LDS by the Tensor Data Mover (wave 0), with the
//           TDM pad feature providing the +2-DWORD/row bank-conflict padding.
// ---------------------------------------------------------------------------
__global__ __launch_bounds__(128)
void mhc_gate_kernel(const float* __restrict__ x,
                     const float* __restrict__ w_pre, const float* __restrict__ w_post,
                     const float* __restrict__ w_res,
                     const float* __restrict__ b_pre, const float* __restrict__ b_res,
                     const float* __restrict__ b_post,
                     const float* __restrict__ alpha_pre, const float* __restrict__ alpha_res,
                     const float* __restrict__ alpha_post,
                     const float* __restrict__ rms_w,
                     float* __restrict__ out,
                     bf16*  __restrict__ xpre_ws,
                     float* __restrict__ hpost_ws)
{
    extern __shared__ char smem[];
    float* x_lds = (float*)smem;                 // [SB][XS]
    float* scale = x_lds + SB * XS;              // [SB]
    float* rmsw  = scale + SB;                   // [256]
    float* hst   = rmsw + 256;                   // [SB][HS]
    bf16*  wcat  = (bf16*)(hst + SB * HS);       // [32][256]

    const int tid  = threadIdx.x;
    const int lane = tid & 31;
    const int wave = tid >> 5;
    const int mrow = lane & 15;     // WMMA matrix row / B column index
    const int hi   = lane >> 4;     // K sub-range selector
    const long pix0 = (long)blockIdx.x * SB;

#ifdef HAVE_TDM
    // --- Tensor Data Mover: DMA the 128x256 fp32 tile HBM -> LDS (wave 0) ---
    if (wave == 0) {
        const unsigned lds_off = (unsigned)(unsigned long long)(const void*)x_lds;
        const unsigned long long ga = (unsigned long long)(const void*)(x + pix0 * SC);
        // D# group 0: count=1 | lds_addr | global_addr | type=2
        u32x4 g0;
        g0[0] = 1u;
        g0[1] = lds_off;
        g0[2] = (unsigned)(ga & 0xFFFFFFFFull);
        g0[3] = (unsigned)((ga >> 32) & 0x1FFFFFFull) | (2u << 30);
        // D# group 1: data_size=4B(2), pad_enable, pad_interval=7 (256 DW),
        //             pad_amount=1 (2 DW) -> LDS row stride 258 floats.
        // tensor_dim0=256, tensor_dim1=PX_TOTAL, tile_dim0=256, tile_dim1=SB,
        // tensor_dim0_stride=256.
        i32x8 g1;
        g1[0] = (int)((2u << 16) | (1u << 20) | (7u << 22) | (1u << 25));
        g1[1] = (int)(256u << 16);                               // dim0 lo16 << 16
        g1[2] = (int)(((unsigned)PX_TOTAL & 0xFFFFu) << 16);     // dim0 hi16 | dim1 lo16
        g1[3] = (int)(((unsigned)PX_TOTAL >> 16) | (256u << 16)); // dim1 hi16 | tile_dim0
        g1[4] = (int)(SB);                                       // tile_dim1 | tile_dim2=0
        g1[5] = (int)256;                                        // dim0_stride lo32
        g1[6] = 0;                                               // stride hi | dim1_stride lo
        g1[7] = 0;
        i32x4 g2 = {0, 0, 0, 0};
        i32x4 g3 = {0, 0, 0, 0};
#if __clang_major__ >= 23
        i32x8 g4 = {0, 0, 0, 0, 0, 0, 0, 0};
        __builtin_amdgcn_tensor_load_to_lds(g0, g1, g2, g3, g4, 0);
#else
        __builtin_amdgcn_tensor_load_to_lds(g0, g1, g2, g3, 0);
#endif
    }
#endif

    // rms weights (overlapped with TDM)
    for (int i = tid; i < 256; i += 128) rmsw[i] = rms_w[i];
    // concat weights [pre(4) | post(4) | res(16) | zero-pad(8)] -> bf16 LDS
    for (int i = tid; i < 32 * 256; i += 128) {
        int n = i >> 8, k = i & 255;
        float v = 0.f;
        if      (n < 4)  v = w_pre [n * 256 + k];
        else if (n < 8)  v = w_post[(n - 4) * 256 + k];
        else if (n < 24) v = w_res [(n - 8) * 256 + k];
        wcat[n * 256 + k] = (bf16)v;
    }

#ifdef HAVE_TDM
    if (wave == 0) __builtin_amdgcn_s_wait_tensorcnt(0);
#else
    // fallback: stage x tile with coalesced float4 copies
    {
        const float4* xg = (const float4*)(x + pix0 * SC);
        for (int i = tid; i < SB * 64; i += 128) {
            int p = i >> 6, q = i & 63;
            float4 v = xg[p * 64 + q];
            float* dst = x_lds + p * XS + q * 4;
            dst[0] = v.x; dst[1] = v.y; dst[2] = v.z; dst[3] = v.w;
        }
    }
#endif
    __syncthreads();

    // per-pixel RMS scale (one pixel per thread)
    {
        const float* xr = x_lds + tid * XS;
        float s = 0.f;
        for (int k = 0; k < 256; ++k) { float v = xr[k]; s += v * v; }
        scale[tid] = rsqrtf(s * (1.0f / 256.0f) + 1.1920929e-7f);
    }
    __syncthreads();

    // GEMM: each wave -> 32 pixels = 2 M-tiles; N tiles: outputs 0-15 / 16-31
    for (int mt = 0; mt < 2; ++mt) {
        const int mbase = wave * 32 + mt * 16;
        const int p = mbase + mrow;
        const float* xr = x_lds + p * XS;
        const float sc = scale[p];
        f32x8 acc0 = {}; f32x8 acc1 = {};
        for (int kc = 0; kc < 8; ++kc) {
            const int k0 = kc * 32 + hi * 8;
            bf16x16 a;
#pragma unroll
            for (int i = 0; i < 8; ++i) {
                int ka = k0 + i, kb = k0 + 16 + i;
                a[i]     = (bf16)(xr[ka] * sc * rmsw[ka]);
                a[8 + i] = (bf16)(xr[kb] * sc * rmsw[kb]);
            }
            const int kr = kc * 32 + hi * 8;
            const bf16* w0 = wcat + (mrow)      * 256 + kr;
            const bf16* w1 = wcat + (16 + mrow) * 256 + kr;
            bf16x16 bfr0, bfr1;
#pragma unroll
            for (int i = 0; i < 16; ++i) { int o = i < 8 ? i : 8 + i; bfr0[i] = w0[o]; bfr1[i] = w1[o]; }
            acc0 = __builtin_amdgcn_wmma_f32_16x16x32_bf16(false, a, false, bfr0, (short)0, acc0, false, false);
            acc1 = __builtin_amdgcn_wmma_f32_16x16x32_bf16(false, a, false, bfr1, (short)0, acc1, false, false);
        }
#pragma unroll
        for (int r = 0; r < 8; ++r) {
            int pp = mbase + r + 8 * hi;
            hst[pp * HS + mrow]      = acc0[r];
            hst[pp * HS + 16 + mrow] = acc1[r];
        }
    }
    __syncthreads();

    // per-pixel: gates, Sinkhorn, mixes (one pixel per thread)
    {
        const int p = tid;
        const long pg = pix0 + p;
        const float ap = alpha_pre[0], ar = alpha_res[0], apo = alpha_post[0];
        const float* hr = hst + p * HS;
        float hpre[4], hpost[4], hres[16];
#pragma unroll
        for (int s = 0; s < 4; ++s) {
            hpre[s]  = 1.0f / (1.0f + __expf(-(ap  * hr[s]     + b_pre[s])));
            hpost[s] = 2.0f / (1.0f + __expf(-(apo * hr[4 + s] + b_post[s])));
        }
        float m = -3.402823466e38f;
#pragma unroll
        for (int i = 0; i < 16; ++i) { hres[i] = ar * hr[8 + i] + b_res[i]; m = fmaxf(m, hres[i]); }
#pragma unroll
        for (int i = 0; i < 16; ++i) hres[i] = __expf(hres[i] - m);
        for (int it = 0; it < 20; ++it) {
#pragma unroll
            for (int t = 0; t < 4; ++t) {   // column normalize (axis -2)
                float inv = 1.0f / (hres[t] + hres[4 + t] + hres[8 + t] + hres[12 + t] + 1e-6f);
                hres[t] *= inv; hres[4 + t] *= inv; hres[8 + t] *= inv; hres[12 + t] *= inv;
            }
#pragma unroll
            for (int s = 0; s < 4; ++s) {   // row normalize (axis -1)
                float inv = 1.0f / (hres[4*s] + hres[4*s+1] + hres[4*s+2] + hres[4*s+3] + 1e-6f);
                hres[4*s] *= inv; hres[4*s+1] *= inv; hres[4*s+2] *= inv; hres[4*s+3] *= inv;
            }
        }
#pragma unroll
        for (int s = 0; s < 4; ++s) hpost_ws[pg * 4 + s] = hpost[s];

        const float* xr = x_lds + p * XS;
        float* og = out + pg * 256;
        bf16*  xp = xpre_ws + pg * 64;
        for (int c = 0; c < 64; ++c) {
            float x0 = xr[c], x1 = xr[64 + c], x2 = xr[128 + c], x3 = xr[192 + c];
            xp[c] = (bf16)(hpre[0]*x0 + hpre[1]*x1 + hpre[2]*x2 + hpre[3]*x3);
            og[c]        = hres[0]*x0  + hres[1]*x1  + hres[2]*x2  + hres[3]*x3;
            og[64 + c]   = hres[4]*x0  + hres[5]*x1  + hres[6]*x2  + hres[7]*x3;
            og[128 + c]  = hres[8]*x0  + hres[9]*x1  + hres[10]*x2 + hres[11]*x3;
            og[192 + c]  = hres[12]*x0 + hres[13]*x1 + hres[14]*x2 + hres[15]*x3;
        }
    }
}

// ---------------------------------------------------------------------------
// Kernel B: 3x3 conv (C=64->64) as implicit GEMM (bf16 WMMA, K=576) on a
//           16x16 spatial tile with halo, fused h_post-gated RMW into out.
// ---------------------------------------------------------------------------
__global__ __launch_bounds__(128)
void mhc_conv_kernel(const bf16*  __restrict__ xpre_ws,
                     const float* __restrict__ conv_w,
                     const float* __restrict__ hpost_ws,
                     float* __restrict__ out)
{
    extern __shared__ char smem[];
    bf16* wl  = (bf16*)smem;                  // transposed weights [64][WS_CONV]
    bf16* inl = wl + 64 * WS_CONV;            // halo tile [324][68]

    const int tid  = threadIdx.x;
    const int lane = tid & 31;
    const int wave = tid >> 5;
    const int mrow = lane & 15;   // A row (= lx) / B column (= co low bits)
    const int hi   = lane >> 4;

    const int blk = blockIdx.x;               // 16 * 8 * 8 tiles
    const int b  = blk >> 6;
    const int ty = (blk >> 3) & 7;
    const int tx = blk & 7;

    // weights: conv_w flat [(dy*3+dx)*64+ci][co] -> wl[co][k] (bf16, padded)
    for (int i = tid; i < 576 * 64; i += 128) {
        int k = i >> 6, co = i & 63;
        wl[co * WS_CONV + k] = (bf16)conv_w[i];
    }
    // halo tile: 18x18 pixel rows of 64 bf16 (copied as 32 dword pairs),
    // zero-filled at image borders (SAME padding).
    {
        unsigned int* inlu = (unsigned int*)inl;
        const unsigned int* xg = (const unsigned int*)xpre_ws;
        for (int i = tid; i < 324 * 32; i += 128) {
            int q = i >> 5, pr = i & 31;
            int hy = q / 18, hx = q % 18;
            int gy = ty * 16 + hy - 1, gx = tx * 16 + hx - 1;
            unsigned int v = 0u;
            if (gy >= 0 && gy < 128 && gx >= 0 && gx < 128) {
                long g = ((long)b * 128 + gy) * 128 + gx;
                v = xg[g * 32 + pr];
            }
            inlu[q * 34 + pr] = v;
        }
    }
    __syncthreads();

    // each wave: 4 consecutive row-tiles (ly = mt), 4 N-tiles of 16 channels
    for (int t = 0; t < 4; ++t) {
        const int mt = wave * 4 + t;          // = local output row ly
        f32x8 acc[4] = {{}, {}, {}, {}};
        for (int kc = 0; kc < 18; ++kc) {
            const int tap = kc >> 1;
            const int dy = tap / 3, dx = tap % 3;
            const int cb = (kc & 1) * 32;
            const bf16* arow = inl + ((mt + dy) * 18 + (mrow + dx)) * 68 + cb + hi * 8;
            bf16x16 a;
#pragma unroll
            for (int i = 0; i < 8; ++i) { a[i] = arow[i]; a[8 + i] = arow[16 + i]; }
#pragma unroll
            for (int nt = 0; nt < 4; ++nt) {
                const bf16* wrow = wl + (nt * 16 + mrow) * WS_CONV + kc * 32 + hi * 8;
                bf16x16 bfr;
#pragma unroll
                for (int i = 0; i < 16; ++i) bfr[i] = wrow[i < 8 ? i : 8 + i];
                acc[nt] = __builtin_amdgcn_wmma_f32_16x16x32_bf16(false, a, false, bfr, (short)0, acc[nt], false, false);
            }
        }
        // fused epilogue: out[g, s, co] += h_post[g, s] * layer_out[g, co]
        const long gb = ((long)b * 128 + ty * 16 + mt) * 128 + tx * 16;
#pragma unroll
        for (int r = 0; r < 8; ++r) {
            const long g = gb + r + 8 * hi;
            __builtin_prefetch(out + g * 256, 1, 3);
            const float hp0 = hpost_ws[g * 4 + 0];
            const float hp1 = hpost_ws[g * 4 + 1];
            const float hp2 = hpost_ws[g * 4 + 2];
            const float hp3 = hpost_ws[g * 4 + 3];
            float* ob = out + g * 256 + mrow;
#pragma unroll
            for (int nt = 0; nt < 4; ++nt) {
                const float v = acc[nt][r];
                const int co = nt * 16;
                ob[co]        += hp0 * v;
                ob[64 + co]   += hp1 * v;
                ob[128 + co]  += hp2 * v;
                ob[192 + co]  += hp3 * v;
            }
        }
    }
}

// ---------------------------------------------------------------------------
extern "C" void kernel_launch(void* const* d_in, const int* in_sizes, int n_in,
                              void* d_out, int out_size, void* d_ws, size_t ws_size,
                              hipStream_t stream)
{
    const float* x          = (const float*)d_in[0];
    const float* w_pre      = (const float*)d_in[1];
    const float* w_post     = (const float*)d_in[2];
    const float* w_res      = (const float*)d_in[3];
    const float* b_pre      = (const float*)d_in[4];
    const float* b_res      = (const float*)d_in[5];
    const float* b_post     = (const float*)d_in[6];
    const float* alpha_pre  = (const float*)d_in[7];
    const float* alpha_res  = (const float*)d_in[8];
    const float* alpha_post = (const float*)d_in[9];
    const float* rms_w      = (const float*)d_in[10];
    const float* conv_w     = (const float*)d_in[11];
    float* out = (float*)d_out;

    bf16*  xpre  = (bf16*)d_ws;                                       // 32 MB
    float* hpost = (float*)((char*)d_ws + (size_t)PX_TOTAL * 64 * 2); //  4 MB

    const size_t ldsA = (size_t)(SB * XS + SB + 256 + SB * HS) * 4 + 32 * 256 * 2;
    const size_t ldsB = (size_t)64 * WS_CONV * 2 + (size_t)324 * 68 * 2;

    mhc_gate_kernel<<<dim3(PX_TOTAL / SB), dim3(128), ldsA, stream>>>(
        x, w_pre, w_post, w_res, b_pre, b_res, b_post,
        alpha_pre, alpha_res, alpha_post, rms_w, out, xpre, hpost);

    mhc_conv_kernel<<<dim3(16 * 8 * 8), dim3(128), ldsB, stream>>>(
        xpre, conv_w, hpost, out);
}
